// WGCN_71854802862665
// MI455X (gfx1250) — compile-verified, matrix operand order checked by reading, not run on hardware
//
#include <hip/hip_runtime.h>

// WGCN forward on MI455X (gfx1250, wave32).
// Big GEMM (grpha @ emb): v_wmma_f32_16x16x32_bf16, A converted f32->bf16 in-flight,
//   K-split across workgroups for machine fill, NT loads for the streamed grpha.
// Small GEMMs + qos: v_wmma_f32_16x16x4_f32 (exact fp32).
// pack transpose uses CDNA5 async global->LDS (ASYNCcnt) data movement.

#define U_CNT 2048
#define S_CNT 8192
#define N_TOT 10240
#define DD    128

typedef __attribute__((ext_vector_type(16))) __bf16 v16bf;
typedef __attribute__((ext_vector_type(8)))  __bf16 v8bf;
typedef __attribute__((ext_vector_type(8)))  float  v8f;
typedef __attribute__((ext_vector_type(4)))  float  v4f;
typedef __attribute__((ext_vector_type(2)))  float  v2f;

// B fragment for 16x16x32 bf16 WMMA: per lane (n = lane%16, half = lane/16),
// elements 0..7  = B[k][n], k = half*8 .. half*8+7       (16 contiguous bytes)
// elements 8..15 = B[k][n], k = 16+half*8 .. 16+half*8+7 (16 contiguous bytes)
__device__ inline v16bf load_b_frag(const __bf16* __restrict__ p) {
  v8bf lo = *(const v8bf*)(p);
  v8bf hi = *(const v8bf*)(p + 16);
  v16bf r;
#pragma unroll
  for (int i = 0; i < 8; ++i) { r[i] = lo[i]; r[8 + i] = hi[i]; }
  return r;
}

// ---------------------------------------------------------------------------
// init: X = concat(user_emb, item_emb); SUM = X
// ---------------------------------------------------------------------------
__global__ __launch_bounds__(256) void init_kernel(const float* __restrict__ ue,
                                                   const float* __restrict__ ie,
                                                   float* __restrict__ X,
                                                   float* __restrict__ SUM) {
  size_t i = (size_t)blockIdx.x * 256 + threadIdx.x;     // < N_TOT*DD
  size_t row = i >> 7;
  float v = (row < U_CNT) ? ue[i] : ie[i - (size_t)U_CNT * DD];
  X[i]   = v;
  SUM[i] = v;
}

// ---------------------------------------------------------------------------
// pack: XT[col][row] = bf16(X[row][col])  (tiled transpose; tile is staged with
// CDNA5 async global->LDS DMA, tracked by ASYNCcnt).
// The single __shared__ array is at LDS offset 0, so the raw LDS byte offset
// passed as VDST is just the element offset.
// ---------------------------------------------------------------------------
__global__ void pack_kernel(const float* __restrict__ X, __bf16* __restrict__ XT) {
  __shared__ float tile[16][17];
  int cb = blockIdx.x * 16;   // column base (0..112)
  int rb = blockIdx.y * 16;   // row base    (0..10224)
  int tx = threadIdx.x, ty = threadIdx.y;

  unsigned lds_off = (unsigned)((ty * 17 + tx) * 4);   // &tile[ty][tx] - LDS base
  unsigned long long gaddr =
      (unsigned long long)(uintptr_t)(X + (size_t)(rb + ty) * DD + cb + tx);
  asm volatile("global_load_async_to_lds_b32 %0, %1, off"
               :: "v"(lds_off), "v"(gaddr)
               : "memory");
  asm volatile("s_wait_asynccnt 0x0" ::: "memory");
  __syncthreads();

  XT[(size_t)(cb + ty) * N_TOT + rb + tx] = (__bf16)tile[tx][ty];
}

// ---------------------------------------------------------------------------
// propagate (partial): PP[kc] = grpha[:, k-range] @ X[k-range, :]  (unscaled)
// one wave = 16 output rows x 128 cols (8 col-tiles) x one K-chunk.
// grpha is streamed with non-temporal loads (read-once, 419 MB) so it does not
// evict the L2-resident XT/X/Y/SUM working set. K-loop unrolled x2 so loads of
// the next step issue under the current WMMA chain.
// ---------------------------------------------------------------------------
__global__ __launch_bounds__(256) void prop_kernel(const float* __restrict__ G,
                                                   const __bf16* __restrict__ XT,
                                                   float* __restrict__ PP,
                                                   int chunk) {
  const int lane = threadIdx.x & 31;
  const int wave = threadIdx.x >> 5;
  const int row0 = (blockIdx.x * 8 + wave) * 16;
  const int kc   = blockIdx.y;
  const int k0   = kc * chunk;
  const int m    = lane & 15;
  const int half = lane >> 4;

  const float* __restrict__ arow = G + (size_t)(row0 + m) * N_TOT;

  v8f acc[8] = {};

#pragma unroll 2
  for (int kb = k0; kb < k0 + chunk; kb += 32) {
    // A fragment: lane holds grpha[row0+m][kb + {half*8..+7, 16+half*8..+7}]
    v4f a0 = __builtin_nontemporal_load((const v4f*)(arow + kb + half * 8));
    v4f a1 = __builtin_nontemporal_load((const v4f*)(arow + kb + half * 8 + 4));
    v4f a2 = __builtin_nontemporal_load((const v4f*)(arow + kb + 16 + half * 8));
    v4f a3 = __builtin_nontemporal_load((const v4f*)(arow + kb + 16 + half * 8 + 4));
    v16bf bfr[8];
#pragma unroll
    for (int nt = 0; nt < 8; ++nt) {
      bfr[nt] = load_b_frag(XT + (size_t)(nt * 16 + m) * N_TOT + kb + half * 8);
    }
    v16bf af;
#pragma unroll
    for (int j = 0; j < 4; ++j) {
      af[j]      = (__bf16)a0[j];
      af[4 + j]  = (__bf16)a1[j];
      af[8 + j]  = (__bf16)a2[j];
      af[12 + j] = (__bf16)a3[j];
    }
#pragma unroll
    for (int nt = 0; nt < 8; ++nt) {
      acc[nt] = __builtin_amdgcn_wmma_f32_16x16x32_bf16(
          false, af, false, bfr[nt], (short)0, acc[nt], false, false);
    }
  }

  float* __restrict__ out = PP + (size_t)kc * N_TOT * DD;
#pragma unroll
  for (int nt = 0; nt < 8; ++nt) {
#pragma unroll
    for (int r = 0; r < 8; ++r) {
      out[(size_t)(row0 + r + 8 * half) * DD + nt * 16 + m] = acc[nt][r];
    }
  }
}

// Y = scale * sum_kc PP[kc]
__global__ __launch_bounds__(256) void prop_reduce_kernel(const float* __restrict__ PP,
                                                          float* __restrict__ Y,
                                                          int ksplit, float scale) {
  size_t i = (size_t)blockIdx.x * 256 + threadIdx.x;
  const size_t NEL = (size_t)N_TOT * DD;
  float s = 0.f;
  for (int k = 0; k < ksplit; ++k) s += PP[(size_t)k * NEL + i];
  Y[i] = s * scale;
}

// ---------------------------------------------------------------------------
// transform step: out = act?( X @ W.T + b ) : X @ W.T + b ; optional SUM += out
// exact fp32 via v_wmma_f32_16x16x4_f32. One wave = 16 rows x 4 col-tiles
// (column-split by 2 for machine fill; A re-reads hit L2-resident X).
// ---------------------------------------------------------------------------
__global__ __launch_bounds__(256) void gemm128_kernel(const float* __restrict__ X,
                                                      const float* __restrict__ W,
                                                      const float* __restrict__ bias,
                                                      float* __restrict__ out,
                                                      float* __restrict__ sum,
                                                      int do_act) {
  const int lane = threadIdx.x & 31;
  const int wave = threadIdx.x >> 5;
  const int gid  = blockIdx.x * 8 + wave;   // 0..1279
  const int row0 = (gid >> 1) * 16;
  const int colh = gid & 1;                 // which 4 column tiles
  const int m    = lane & 15;
  const int half = lane >> 4;

  const float* __restrict__ xrow = X + (size_t)(row0 + m) * DD;

  v8f acc[4] = {};

  for (int kb = 0; kb < DD; kb += 4) {
    v2f a = *(const v2f*)(xrow + kb + 2 * half);
    v2f b[4];
#pragma unroll
    for (int t = 0; t < 4; ++t) {
      // B[k][n] = W[n][k] : lane loads W[nt*16+m][kb+2*half+{0,1}]
      int nt = colh * 4 + t;
      b[t] = *(const v2f*)(W + (size_t)(nt * 16 + m) * DD + kb + 2 * half);
    }
#pragma unroll
    for (int t = 0; t < 4; ++t) {
      acc[t] = __builtin_amdgcn_wmma_f32_16x16x4_f32(
          false, a, false, b[t], (short)0, acc[t], false, false);
    }
  }

#pragma unroll
  for (int t = 0; t < 4; ++t) {
    int nt = colh * 4 + t;
    float bv = bias[nt * 16 + m];
#pragma unroll
    for (int r = 0; r < 8; ++r) {
      float v = acc[t][r] + bv;
      if (do_act) v = (v > 0.f) ? v : 0.01f * v;
      size_t idx = (size_t)(row0 + r + 8 * half) * DD + nt * 16 + m;
      out[idx] = v;
      if (sum) sum[idx] += v;
    }
  }
}

// ---------------------------------------------------------------------------
// fused qos + loss: per 16x16 tile of users@items.T, apply mask and
// weight*|pre - true|, reduce to one partial per block (deterministic).
// tq/lw/mask are streamed (201 MB, read-once) -> non-temporal loads.
// ---------------------------------------------------------------------------
__global__ __launch_bounds__(256) void loss_kernel(const float* __restrict__ SUM,
                                                   const float* __restrict__ tq,
                                                   const float* __restrict__ lw,
                                                   const int* __restrict__ mask,
                                                   float* __restrict__ partial) {
  const int lane = threadIdx.x & 31;
  const int wave = threadIdx.x >> 5;
  const int tile = blockIdx.x * 8 + wave;   // 0..65535
  const int ut = tile >> 9;                 // user tile  0..127
  const int st = tile & 511;                // item tile  0..511
  const int m    = lane & 15;
  const int half = lane >> 4;

  const float* __restrict__ urow = SUM + (size_t)(ut * 16 + m) * DD;
  const float* __restrict__ irow = SUM + (size_t)(U_CNT + st * 16 + m) * DD;

  v8f acc = {};
  for (int kb = 0; kb < DD; kb += 4) {
    v2f a = *(const v2f*)(urow + kb + 2 * half);
    v2f b = *(const v2f*)(irow + kb + 2 * half);
    acc = __builtin_amdgcn_wmma_f32_16x16x4_f32(
        false, a, false, b, (short)0, acc, false, false);
  }

  float local = 0.f;
#pragma unroll
  for (int r = 0; r < 8; ++r) {
    int u = ut * 16 + r + 8 * half;
    int s = st * 16 + m;
    size_t idx = (size_t)u * S_CNT + s;
    float pre = acc[r] * (float)__builtin_nontemporal_load(mask + idx);
    float t   = __builtin_nontemporal_load(tq + idx);
    float w   = __builtin_nontemporal_load(lw + idx);
    local += w * fabsf(pre - t);
  }
#pragma unroll
  for (int off = 16; off > 0; off >>= 1) local += __shfl_xor(local, off, 32);

  __shared__ float wsum[8];
  if (lane == 0) wsum[wave] = local;
  __syncthreads();
  if (threadIdx.x == 0) {
    float t = 0.f;
#pragma unroll
    for (int i = 0; i < 8; ++i) t += wsum[i];
    partial[blockIdx.x] = t;
  }
}

__global__ __launch_bounds__(256) void reduce_kernel(const float* __restrict__ partial,
                                                     int n, float* __restrict__ out) {
  __shared__ float sm[256];
  float s = 0.f;
  for (int i = threadIdx.x; i < n; i += 256) s += partial[i];
  sm[threadIdx.x] = s;
  __syncthreads();
  for (int off = 128; off > 0; off >>= 1) {
    if ((int)threadIdx.x < off) sm[threadIdx.x] += sm[threadIdx.x + off];
    __syncthreads();
  }
  if (threadIdx.x == 0) out[0] = sm[0];
}

// ---------------------------------------------------------------------------
extern "C" void kernel_launch(void* const* d_in, const int* in_sizes, int n_in,
                              void* d_out, int out_size, void* d_ws, size_t ws_size,
                              hipStream_t stream) {
  (void)in_sizes; (void)n_in; (void)out_size;
  const float* user_emb = (const float*)d_in[0];
  const float* item_emb = (const float*)d_in[1];
  const float* grpha    = (const float*)d_in[2];
  const float* W1 = (const float*)d_in[3];
  const float* b1 = (const float*)d_in[4];
  const float* W2 = (const float*)d_in[5];
  const float* b2 = (const float*)d_in[6];
  const float* W3 = (const float*)d_in[7];
  const float* b3 = (const float*)d_in[8];
  const float* tq = (const float*)d_in[9];
  const float* lw = (const float*)d_in[10];
  const int* mask = (const int*)d_in[11];
  float* out = (float*)d_out;

  char* ws = (char*)d_ws;
  const size_t EMB = (size_t)N_TOT * DD * sizeof(float);          // 5,242,880 B
  const size_t XTB = (size_t)N_TOT * DD * 2;                      // 2,621,440 B
  const int NPART  = (U_CNT / 16) * (S_CNT / 16) / 8;             // 8192 partials

  float*  X    = (float*)(ws + 0 * EMB);
  float*  Y    = (float*)(ws + 1 * EMB);
  float*  H1   = (float*)(ws + 2 * EMB);
  float*  SUM  = (float*)(ws + 3 * EMB);
  __bf16* XT   = (__bf16*)(ws + 4 * EMB);
  float*  PART = (float*)(ws + 4 * EMB + XTB);
  char*   ppb  = ws + 4 * EMB + XTB + ((NPART * sizeof(float) + 255) & ~(size_t)255);
  float*  PP   = (float*)ppb;

  // K-split factor for the big GEMM, limited by available scratch (graceful).
  const size_t base = (size_t)(ppb - ws);
  int ksplit = 1;
  if (ws_size >= base + 8 * EMB)      ksplit = 8;
  else if (ws_size >= base + 4 * EMB) ksplit = 4;
  else if (ws_size >= base + 2 * EMB) ksplit = 2;
  const int chunk = N_TOT / ksplit;   // multiple of 64 for all choices

  init_kernel<<<(N_TOT * DD) / 256, 256, 0, stream>>>(user_emb, item_emb, X, SUM);
  for (int layer = 0; layer < 3; ++layer) {
    pack_kernel<<<dim3(DD / 16, N_TOT / 16), dim3(16, 16), 0, stream>>>(X, XT);
    prop_kernel<<<dim3(N_TOT / 128, ksplit), 256, 0, stream>>>(grpha, XT, PP, chunk);
    prop_reduce_kernel<<<(N_TOT * DD) / 256, 256, 0, stream>>>(
        PP, Y, ksplit, 1.0f / (float)(1 + layer));
    gemm128_kernel<<<N_TOT / 64, 256, 0, stream>>>(Y,  W1, b1, H1, nullptr, 1);
    gemm128_kernel<<<N_TOT / 64, 256, 0, stream>>>(H1, W2, b2, Y,  nullptr, 1);
    gemm128_kernel<<<N_TOT / 64, 256, 0, stream>>>(Y,  W3, b3, X,  SUM,     0);
  }
  loss_kernel<<<NPART, 256, 0, stream>>>(SUM, tq, lw, mask, PART);
  reduce_kernel<<<1, 256, 0, stream>>>(PART, NPART, out);
}
